// FNO4d_24747601559618
// MI455X (gfx1250) — matrix operand order repaired and checked by reading, not compiled, not run on hardware
//
#include <hip/hip_runtime.h>
#include <hip/hip_bf16.h>
#include <math.h>

typedef __attribute__((ext_vector_type(16))) _Float16 v16h;
typedef __attribute__((ext_vector_type(8)))  float    v8f;

#define PTS 786432L   // 64*64*64*3 grid points
#define NCH 32

// ---------------- Async global->LDS staging (CDNA5, ASYNCcnt-tracked) -------
#if defined(__has_builtin)
#if __has_builtin(__builtin_amdgcn_global_load_async_to_lds_b128)
#define HAVE_ASYNC_LDS 1
#endif
#endif

#ifdef HAVE_ASYNC_LDS
typedef int v4i_ __attribute__((vector_size(16)));
typedef v4i_ __attribute__((address_space(1))) gas_v4i;   // global
typedef v4i_ __attribute__((address_space(3))) las_v4i;   // LDS

__device__ __forceinline__ void async_copy16(const void* g, void* l) {
    __builtin_amdgcn_global_load_async_to_lds_b128(
        (gas_v4i*)(void*)g, (las_v4i*)l, 0, 0);
}
__device__ __forceinline__ void async_wait0() {
#if __has_builtin(__builtin_amdgcn_s_wait_asynccnt)
    __builtin_amdgcn_s_wait_asynccnt(0);
#else
    asm volatile("s_wait_asynccnt 0" ::: "memory");
#endif
}
#endif

// ---------------- Branchless exact-grade GELU -------------------------------
// erf via Abramowitz-Stegun 7.1.26 (|err|<1.5e-7), no exec-mask divergence:
// rcp + 5 fma + fast exp -> keeps WMMA loops free of saveexec sequences.
__device__ __forceinline__ float gelu_exact(float v) {
    float x  = v * 0.70710678118f;
    float ax = fabsf(x);
    float t  = __builtin_amdgcn_rcpf(1.0f + 0.3275911f * ax);
    float poly = t * (0.254829592f +
                 t * (-0.284496736f +
                 t * (1.421413741f +
                 t * (-1.453152027f +
                 t * 1.061405429f))));
    float e = __expf(-x * x);
    float erf_ax = 1.0f - poly * e;
    float er = copysignf(erf_ax, x);
    return 0.5f * v * (1.0f + er);
}

// ---------------- WMMA fragment loaders (per CDNA5 ISA 7.12.2 layouts) ------
// A (16x32 f16, M x K): lane l -> row M=l%16; half h -> K = h + (h&8) + ((l>=16)?8:0)
__device__ __forceinline__ v16h load_a_w32(const float* __restrict__ Wmat, int obase, int lane) {
    int m = lane & 15;
    int koff = (lane >> 4) << 3;
    v16h a;
#pragma unroll
    for (int h = 0; h < 16; ++h) {
        int k = h + (h & 8) + koff;
        a[h] = (_Float16)Wmat[(obase + m) * 32 + k];
    }
    return a;
}
// B (32x16 f16, K x N) from channel-major activations H[c*P + p]:
// lane l -> col N=l%16; half h -> K = h + (h&8) + ((l>=16)?8:0)
__device__ __forceinline__ v16h load_b_act(const float* __restrict__ H, long pbase, int lane) {
    int n = lane & 15;
    int koff = (lane >> 4) << 3;
    v16h b;
#pragma unroll
    for (int h = 0; h < 16; ++h) {
        long c = (long)(h + (h & 8) + koff);
        b[h] = (_Float16)H[c * PTS + pbase + n];
    }
    return b;
}

// ---------------- fc0: 9 -> 32 lift (grid coords generated on the fly) ------
__global__ void fno_fc0_kernel(const float* __restrict__ x, const float* __restrict__ w,
                               const float* __restrict__ b, float* __restrict__ hout) {
    long p = (long)blockIdx.x * blockDim.x + threadIdx.x;
    long wi = p % 3; long t = p / 3;
    long z = t % 64; t /= 64;
    long y = t % 64; long xx = t / 64;
    float in[9];
#pragma unroll
    for (int j = 0; j < 5; ++j) in[j] = x[p * 5 + j];
    in[5] = (float)xx * (1.0f / 63.0f);
    in[6] = (float)y  * (1.0f / 63.0f);
    in[7] = (float)z  * (1.0f / 63.0f);
    in[8] = (float)wi * 0.5f;
#pragma unroll
    for (int c = 0; c < NCH; ++c) {
        float acc = b[c];
#pragma unroll
        for (int j = 0; j < 9; ++j) acc += w[c * 9 + j] * in[j];
        hout[(long)c * PTS + p] = acc;   // lanes = consecutive p -> coalesced
    }
}

// ---------------- F1: forward truncated DFT over (Z,W) ----------------------
// in : h[c][x][y][z][w] (192 contiguous floats per (c,x,y) row)
// out: A1[c][x][y][kz*2+kw] complex (kz<8, kw<2)
__global__ void fno_f1_kernel(const float* __restrict__ h, float* __restrict__ A1) {
    __shared__ float ld[16][192];        // 12 KB, contiguous match of global rows
    __shared__ float twr[64], twi[64];
    int tid = threadIdx.x;
    if (tid < 64) {
        float a = -6.28318530718f * (float)tid / 64.0f;   // e^{-i...}
        twr[tid] = cosf(a); twi[tid] = sinf(a);
    }
    long rowbase = (long)blockIdx.x * 16;
#ifdef HAVE_ASYNC_LDS
#pragma unroll
    for (int it = 0; it < 3; ++it) {     // 3072 floats = 768 x 16B, 256 lanes x 3
        int fo = (it * 256 + tid) * 4;
        async_copy16(h + rowbase * 192 + fo, &ld[0][0] + fo);
    }
    async_wait0();
#else
    for (int idx = tid; idx < 16 * 192; idx += 256) {
        int r = idx / 192, e = idx % 192;
        ld[r][e] = h[(rowbase + r) * 192 + e];
    }
#endif
    __syncthreads();
    int r = tid >> 4, kzw = tid & 15, kz = kzw >> 1, kw = kzw & 1;
    float accr = 0.f, acci = 0.f;
    for (int z = 0; z < 64; ++z) {
        float s0 = ld[r][z * 3 + 0], s1 = ld[r][z * 3 + 1], s2 = ld[r][z * 3 + 2];
        // W-DFT (length 3): kw=0 -> sum; kw=1 -> e^{-2pi i w/3}
        float cr, ci;
        if (kw == 0) { cr = s0 + s1 + s2; ci = 0.f; }
        else { cr = s0 - 0.5f * (s1 + s2); ci = 0.8660254038f * (s2 - s1); }
        int n = (kz * z) & 63;
        float er = twr[n], ei = twi[n];
        accr += er * cr - ei * ci;
        acci += er * ci + ei * cr;
    }
    long o = (rowbase + r) * 16 + kzw;
    A1[o * 2 + 0] = accr; A1[o * 2 + 1] = acci;
}

// ---------------- F2: forward DFT over Y (keep ky in 0..7, 56..63) ----------
// in : A1[c][x][y][kzw] ; out: A2[c][x][kyi][kzw]
__global__ void fno_f2_kernel(const float* __restrict__ A1, float* __restrict__ A2) {
    __shared__ float lri[2048];          // 1024 complex, interleaved (8 KB)
    __shared__ float twr[64], twi[64];
    int tid = threadIdx.x;
    if (tid < 64) {
        float a = -6.28318530718f * (float)tid / 64.0f;
        twr[tid] = cosf(a); twi[tid] = sinf(a);
    }
    long base = (long)blockIdx.x * 1024;      // blockIdx = c*64 + x
#ifdef HAVE_ASYNC_LDS
#pragma unroll
    for (int it = 0; it < 2; ++it) {
        int fo = (it * 256 + tid) * 4;
        async_copy16(A1 + base * 2 + fo, lri + fo);
    }
    async_wait0();
#else
    for (int idx = tid; idx < 2048; idx += 256) lri[idx] = A1[base * 2 + idx];
#endif
    __syncthreads();
    int kyi = tid >> 4, kzw = tid & 15;
    int ky = (kyi < 8) ? kyi : (kyi + 48);
    float ar = 0.f, ai = 0.f;
    for (int y = 0; y < 64; ++y) {
        float vr = lri[(y * 16 + kzw) * 2], vi = lri[(y * 16 + kzw) * 2 + 1];
        int n = (ky * y) & 63;
        float er = twr[n], ei = twi[n];
        ar += er * vr - ei * vi;
        ai += er * vi + ei * vr;
    }
    long o = (long)blockIdx.x * 256 + tid;    // [c][x][kyi][kzw]
    A2[o * 2] = ar; A2[o * 2 + 1] = ai;
}

// ---------------- F3: forward DFT over X ------------------------------------
// in : A2[c][x][kyi][kzw] ; out: A3[c][kxi][kyi][kzw]
__global__ void fno_f3_kernel(const float* __restrict__ A2, float* __restrict__ A3) {
    __shared__ float twr[64], twi[64];
    int tid = threadIdx.x;
    if (tid < 64) {
        float a = -6.28318530718f * (float)tid / 64.0f;
        twr[tid] = cosf(a); twi[tid] = sinf(a);
    }
    __syncthreads();
    int c = blockIdx.x >> 4, kxi = blockIdx.x & 15;
    int kx = (kxi < 8) ? kxi : (kxi + 48);
    long cbase = (long)c * 64 * 256;
    float ar = 0.f, ai = 0.f;
    for (int x = 0; x < 64; ++x) {
        long idx = cbase + (long)x * 256 + tid;
        float vr = A2[idx * 2], vi = A2[idx * 2 + 1];
        int n = (kx * x) & 63;
        float er = twr[n], ei = twi[n];
        ar += er * vr - ei * vi;
        ai += er * vi + ei * vr;
    }
    long o = (long)c * 4096 + (long)kxi * 256 + tid;
    A3[o * 2] = ar; A3[o * 2 + 1] = ai;
}

// ---------------- Spectral multiply: per-mode 32x32 complex matvec ----------
// mode m = kxi*256 + kyi*16 + kzw ; quadrant uniform per 32-mode wave
__global__ void fno_spec_kernel(const float* __restrict__ A3, const float* __restrict__ sw,
                                float* __restrict__ B3) {
    int lane = threadIdx.x & 31, wid = threadIdx.x >> 5;
    int m = ((blockIdx.x * (blockDim.x >> 5)) + wid) * 32 + lane;
    int kxi = m >> 8, kyi = (m >> 4) & 15, kzw = m & 15;
    int q = ((kxi >= 8) ? 1 : 0) + ((kyi >= 8) ? 2 : 0);
    int mx = kxi & 7, my = kyi & 7;
    int wm = (mx * 8 + my) * 16 + kzw;               // 0..1023 within quadrant
    const float* wq = sw + (long)q * (32L * 32L * 1024L * 2L);
    float accr[32], acci[32];
#pragma unroll
    for (int o = 0; o < 32; ++o) { accr[o] = 0.f; acci[o] = 0.f; }
    for (int i = 0; i < 32; ++i) {
        long ii = (long)i * 4096 + m;
        float xr = A3[ii * 2], xi = A3[ii * 2 + 1];
        const float* wrow = wq + (long)i * (32L * 1024L * 2L);
        __builtin_prefetch(wrow + (long)wm * 2, 0, 1);
#pragma unroll 8
        for (int o = 0; o < 32; ++o) {
            const float* wp = wrow + ((long)o * 1024 + wm) * 2;  // lanes contiguous
            float wr = wp[0], wi2 = wp[1];
            accr[o] += wr * xr - wi2 * xi;
            acci[o] += wr * xi + wi2 * xr;
        }
    }
#pragma unroll
    for (int o = 0; o < 32; ++o) {
        long oo = (long)o * 4096 + m;
        B3[oo * 2] = accr[o]; B3[oo * 2 + 1] = acci[o];
    }
}

// ---------------- I1: inverse DFT over X (64 from 16 modes) -----------------
// in : B3[c][kxi][kyi][kzw] ; out: B2[c][x][kyi][kzw]
__global__ void fno_i1_kernel(const float* __restrict__ B3, float* __restrict__ B2) {
    __shared__ float twr[64], twi[64];
    int tid = threadIdx.x;
    if (tid < 64) {
        float a = 6.28318530718f * (float)tid / 64.0f;   // e^{+i...}
        twr[tid] = cosf(a); twi[tid] = sinf(a);
    }
    __syncthreads();
    int c = blockIdx.x >> 6, x = blockIdx.x & 63;
    float ar = 0.f, ai = 0.f;
    for (int kxi = 0; kxi < 16; ++kxi) {
        int kx = (kxi < 8) ? kxi : (kxi + 48);
        long idx = (long)c * 4096 + (long)kxi * 256 + tid;
        float vr = B3[idx * 2], vi = B3[idx * 2 + 1];
        int n = (kx * x) & 63;
        float er = twr[n], ei = twi[n];
        ar += er * vr - ei * vi;
        ai += er * vi + ei * vr;
    }
    long o = (long)blockIdx.x * 256 + tid;    // [c][x][kyi][kzw]
    B2[o * 2] = ar; B2[o * 2 + 1] = ai;
}

// ---------------- I2: inverse DFT over Y ------------------------------------
// in : B2[c][x][kyi][kzw] ; out: B1[c][x][y][kzw]
__global__ void fno_i2_kernel(const float* __restrict__ B2, float* __restrict__ B1) {
    __shared__ float lri[512];           // 256 complex, interleaved (2 KB)
    __shared__ float twr[64], twi[64];
    int tid = threadIdx.x;
    if (tid < 64) {
        float a = 6.28318530718f * (float)tid / 64.0f;
        twr[tid] = cosf(a); twi[tid] = sinf(a);
    }
    long base = (long)blockIdx.x * 256;       // blockIdx = c*64 + x
#ifdef HAVE_ASYNC_LDS
    if (tid < 128)                       // 512 floats = 128 x 16B
        async_copy16(B2 + base * 2 + tid * 4, lri + tid * 4);
    async_wait0();
#else
    lri[tid] = B2[base * 2 + tid];
    lri[256 + tid] = B2[base * 2 + 256 + tid];
#endif
    __syncthreads();
    int kzw = tid & 15, ybase = (tid >> 4) * 4;
#pragma unroll
    for (int dy = 0; dy < 4; ++dy) {
        int y = ybase + dy;
        float ar = 0.f, ai = 0.f;
#pragma unroll
        for (int kyi = 0; kyi < 16; ++kyi) {
            int ky = (kyi < 8) ? kyi : (kyi + 48);
            float vr = lri[(kyi * 16 + kzw) * 2], vi = lri[(kyi * 16 + kzw) * 2 + 1];
            int n = (ky * y) & 63;
            float er = twr[n], ei = twi[n];
            ar += er * vr - ei * vi;
            ai += er * vi + ei * vr;
        }
        long o = (long)blockIdx.x * 1024 + (long)y * 16 + kzw;  // [c][x][y][kzw]
        B1[o * 2] = ar; B1[o * 2 + 1] = ai;
    }
}

// ---------------- I3: inverse DFT over Z + irfft over W ---------------------
// in : B1[c][x][y][kz*2+kw] ; out: h1[c][x][y][z][w] real (192 per row)
__global__ void fno_i3_kernel(const float* __restrict__ B1, float* __restrict__ h1out) {
    __shared__ float lr[16], li[16];
    __shared__ float twr[64], twi[64];
    int tid = threadIdx.x;      // blockDim = 192
    if (tid < 64) {
        float a = 6.28318530718f * (float)tid / 64.0f;
        twr[tid] = cosf(a); twi[tid] = sinf(a);
    }
    if (tid < 32) {
        float v = B1[(long)blockIdx.x * 32 + tid];
        if (tid & 1) li[tid >> 1] = v; else lr[tid >> 1] = v;
    }
    __syncthreads();
    int z = tid / 3, w = tid % 3;
    float g0r = 0.f, g1r = 0.f, g1i = 0.f;
#pragma unroll
    for (int kz = 0; kz < 8; ++kz) {
        int n = (kz * z) & 63;
        float er = twr[n], ei = twi[n];
        g0r += er * lr[kz * 2]     - ei * li[kz * 2];
        g1r += er * lr[kz * 2 + 1] - ei * li[kz * 2 + 1];
        g1i += er * li[kz * 2 + 1] + ei * lr[kz * 2 + 1];
    }
    // irfft length 3: x[w] = (Re g0 + 2 Re(g1 * e^{+2pi i w/3})) / 3
    float e3r = (w == 0) ? 1.0f : -0.5f;
    float e3i = (w == 1) ? 0.8660254038f : ((w == 2) ? -0.8660254038f : 0.0f);
    const float scale = 1.0f / 786432.0f;     // (1/64)^3 * (1/3)
    float val = g0r + 2.0f * (g1r * e3r - g1i * e3i);
    h1out[(long)blockIdx.x * 192 + tid] = val * scale;
}

// ---------------- Combine: h_next = gelu?( h1 + W*h + b ) via WMMA ----------
__global__ void fno_combine_kernel(const float* __restrict__ h, const float* __restrict__ h1,
                                   const float* __restrict__ ww, const float* __restrict__ wb,
                                   float* __restrict__ hout, int doGelu) {
    int lane = threadIdx.x & 31, wid = threadIdx.x >> 5;
    long pbase = ((long)blockIdx.x * (blockDim.x >> 5) + wid) * 16;
    v16h bfrag = load_b_act(h, pbase, lane);
    int n = lane & 15;
    int mo = (lane >> 4) << 3;
#pragma unroll
    for (int t = 0; t < 2; ++t) {
        v16h afrag = load_a_w32(ww, t * 16, lane);
        v8f c;
#pragma unroll
        for (int r = 0; r < 8; ++r) c[r] = 0.0f;
        c = __builtin_amdgcn_wmma_f32_16x16x32_f16(false, afrag, false, bfrag,
                                                   (short)0, c, false, false);
#pragma unroll
        for (int r = 0; r < 8; ++r) {
            int o = t * 16 + r + mo;
            float v = c[r] + wb[o] + h1[(long)o * PTS + pbase + n];
            if (doGelu) v = gelu_exact(v);
            hout[(long)o * PTS + pbase + n] = v;
        }
    }
}

// ---------------- Fused fc1(32->512) + gelu + fc2(512->1) via WMMA ----------
__global__ void fno_fc12_kernel(const float* __restrict__ h, const float* __restrict__ fc1w,
                                const float* __restrict__ fc1b, const float* __restrict__ fc2w,
                                const float* __restrict__ fc2b, float* __restrict__ out) {
    int lane = threadIdx.x & 31, wid = threadIdx.x >> 5;
    long pbase = ((long)blockIdx.x * (blockDim.x >> 5) + wid) * 16;
    v16h bfrag = load_b_act(h, pbase, lane);
    int n = lane & 15;
    int mo = (lane >> 4) << 3;
    float acc = 0.0f;
    for (int t = 0; t < 32; ++t) {          // 32 tiles of 16 hidden units
        v16h afrag = load_a_w32(fc1w, t * 16, lane);
        v8f c;
#pragma unroll
        for (int r = 0; r < 8; ++r) c[r] = 0.0f;
        c = __builtin_amdgcn_wmma_f32_16x16x32_f16(false, afrag, false, bfrag,
                                                   (short)0, c, false, false);
#pragma unroll
        for (int r = 0; r < 8; ++r) {
            int o = t * 16 + r + mo;
            acc += gelu_exact(c[r] + fc1b[o]) * fc2w[o];
        }
    }
    acc += __shfl_xor(acc, 16);             // fold the two row-halves of the column
    if (lane < 16) out[pbase + n] = acc + fc2b[0];
}

// ---------------- Host-side orchestration -----------------------------------
extern "C" void kernel_launch(void* const* d_in, const int* in_sizes, int n_in,
                              void* d_out, int out_size, void* d_ws, size_t ws_size,
                              hipStream_t stream) {
    (void)in_sizes; (void)n_in; (void)out_size; (void)ws_size;
    const float* x     = (const float*)d_in[0];
    const float* fc0w  = (const float*)d_in[1];
    const float* fc0b  = (const float*)d_in[2];
    const float* specw = (const float*)d_in[3];   // (4,4,32,32,8,8,8,2) complex64
    const float* ww    = (const float*)d_in[4];   // (4,32,32)
    const float* wb    = (const float*)d_in[5];   // (4,32)
    const float* fc1w  = (const float*)d_in[6];   // (512,32)
    const float* fc1b  = (const float*)d_in[7];
    const float* fc2w  = (const float*)d_in[8];   // (1,512)
    const float* fc2b  = (const float*)d_in[9];
    float* out = (float*)d_out;

    char* ws = (char*)d_ws;
    const size_t SZH = (size_t)NCH * PTS * sizeof(float);          // 100.7 MB
    float* hA = (float*)(ws);
    float* hB = (float*)(ws + SZH);
    float* h1 = (float*)(ws + 2 * SZH);
    float* A1 = (float*)(ws + 3 * SZH);                            // 16.78 MB (also B1)
    float* A2 = (float*)((char*)A1 + 32L * 64 * 64 * 16 * 2 * 4);  //  4.19 MB (also B2)
    float* A3 = (float*)((char*)A2 + 32L * 64 * 16 * 16 * 2 * 4);  //  1.05 MB
    float* B3 = (float*)((char*)A3 + 32L * 4096 * 2 * 4);          //  1.05 MB

    fno_fc0_kernel<<<3072, 256, 0, stream>>>(x, fc0w, fc0b, hA);

    float* cur = hA; float* nxt = hB;
    for (int layer = 0; layer < 4; ++layer) {
        fno_f1_kernel<<<8192, 256, 0, stream>>>(cur, A1);
        fno_f2_kernel<<<2048, 256, 0, stream>>>(A1, A2);
        fno_f3_kernel<<<512, 256, 0, stream>>>(A2, A3);
        fno_spec_kernel<<<16, 256, 0, stream>>>(A3, specw + (size_t)layer * 8388608, B3);
        fno_i1_kernel<<<2048, 256, 0, stream>>>(B3, A2);
        fno_i2_kernel<<<2048, 256, 0, stream>>>(A2, A1);
        fno_i3_kernel<<<131072, 192, 0, stream>>>(A1, h1);
        fno_combine_kernel<<<6144, 256, 0, stream>>>(cur, h1, ww + layer * 1024,
                                                     wb + layer * 32, nxt,
                                                     (layer < 3) ? 1 : 0);
        float* t = cur; cur = nxt; nxt = t;
    }

    fno_fc12_kernel<<<6144, 256, 0, stream>>>(cur, fc1w, fc1b, fc2w, fc2b, out);
}
